// SelfAttention_77807627535044
// MI455X (gfx1250) — matrix operand order repaired and checked by reading, not compiled, not run on hardware
//
#include <hip/hip_runtime.h>
#include <hip/hip_bf16.h>
#include <math.h>

// ---------------------------------------------------------------------------
// Self-attention forward on gfx1250 (MI455X), bf16 WMMA path.
//   qkv  = x @ w_qkv + b_qkv        (bf16 WMMA GEMM, bf16 out, double-buffered,
//                                    async global->LDS staging when available)
//   attn = flash-attention(q,k,v)   (bf16 WMMA, f32 online softmax)
//   out  = attn @ w_out + b_out     (bf16 WMMA GEMM, f32 out)
//
// Weights are converted to bf16 AND transposed to [n][k] once, so GEMM tile
// staging is contiguous on both A and B sides (16B vectors / async-LDS).
//
// Fragment layouts per CDNA5 ISA 7.12.2 (wave32):
//   A 16x32 (16-bit): lane m = lane&15; K = {0..7}+8*half then {16..23}+8*half
//   B 32x16 (16-bit): lane n = lane&15; K = {0..7}+8*half then {16..23}+8*half
//   C/D 16x16 f32   : lane n = lane&15; VGPR r -> row r + 8*half.
// ---------------------------------------------------------------------------

typedef __attribute__((ext_vector_type(16))) __bf16        v16bf;
typedef __attribute__((ext_vector_type(8)))  float         v8f;
typedef __attribute__((ext_vector_type(4)))  unsigned int  v4u;
typedef __attribute__((ext_vector_type(4)))  int           v4i;

#define AS1 __attribute__((address_space(1)))
#define AS3 __attribute__((address_space(3)))

#if __has_builtin(__builtin_amdgcn_global_load_async_to_lds_b128)
#define HAVE_ASYNC_LDS 1
#else
#define HAVE_ASYNC_LDS 0
#endif

#define DIM    1024
#define NHEAD  16
#define HS     64
#define INNER  1024
#define SEQ    2048
#define BATCH  4
#define ROWS   (BATCH * SEQ)   // 8192
#define QKV_N  (3 * INNER)     // 3072

#define GSTR   40   // GEMM LDS row stride (ushorts) for 32-wide K tiles
#define ASTR   72   // attention LDS row stride (ushorts) for 64-wide tiles

__device__ __forceinline__ unsigned short f2bf(float f) {
  union { float f; unsigned int u; } v; v.f = f;
  unsigned int r = v.u + 0x7FFFu + ((v.u >> 16) & 1u);  // round-to-nearest-even
  return (unsigned short)(r >> 16);
}

union FragBF {                 // 16 bf16 = 8 dwords = two 16B quads
  v16bf v;
  v4u   q[2];
};
union Quad8 { v4u q; unsigned short s[8]; };

// Copy 16B global -> LDS (async when the toolchain exposes the builtin).
__device__ __forceinline__ void copy16_g2l(const unsigned short* g,
                                           unsigned short* l) {
#if HAVE_ASYNC_LDS
  __builtin_amdgcn_global_load_async_to_lds_b128((AS1 v4i*)g, (AS3 v4i*)l,
                                                 0, 0);
#else
  *(v4u*)l = *(const v4u*)g;
#endif
}
__device__ __forceinline__ void wait_async_copies() {
#if HAVE_ASYNC_LDS
#if __has_builtin(__builtin_amdgcn_s_wait_asynccnt)
  __builtin_amdgcn_s_wait_asynccnt(0);
#else
  asm volatile("s_wait_asynccnt 0" ::: "memory");
#endif
#endif
}

// A fragment from row-major [m][k] storage; p -> start of this lane's row.
__device__ __forceinline__ FragBF load_a_frag(const unsigned short* p, int hh) {
  FragBF f;
  f.q[0] = *(const v4u*)(p + 8 * hh);        // K = koff .. koff+7
  f.q[1] = *(const v4u*)(p + 8 * hh + 16);   // K = koff+16 .. koff+23
  return f;
}
// B fragment from [n][k] (column-of-B-contiguous) storage; p -> lane's column.
__device__ __forceinline__ FragBF load_b_frag(const unsigned short* p, int hh) {
  FragBF f;
  f.q[0] = *(const v4u*)(p + 8 * hh);        // K = 8*hh .. +7
  f.q[1] = *(const v4u*)(p + 8 * hh + 16);   // K = 8*hh+16 .. +23
  return f;
}

// ---------------------------------------------------------------------------
__global__ void f32_to_bf16_kernel(const float* __restrict__ in,
                                   unsigned short* __restrict__ out, int n) {
  int i = blockIdx.x * blockDim.x + threadIdx.x;
  if (i < n) out[i] = f2bf(in[i]);
}

// in:[K][N] f32 -> out:[N][K] bf16, 32x32 LDS tiles, 256 threads (32x8).
__global__ void transpose_to_bf16_kernel(const float* __restrict__ in,
                                         unsigned short* __restrict__ out,
                                         int Kdim, int Ndim) {
  __shared__ float tile[32][33];
  const int n0 = blockIdx.x * 32, k0 = blockIdx.y * 32;
  const int tx = threadIdx.x & 31, ty = threadIdx.x >> 5;
  for (int i = 0; i < 32; i += 8)
    tile[ty + i][tx] = in[(size_t)(k0 + ty + i) * Ndim + n0 + tx];
  __syncthreads();
  for (int i = 0; i < 32; i += 8)
    out[(size_t)(n0 + ty + i) * Kdim + k0 + tx] = f2bf(tile[tx][ty + i]);
}

// ---------------------------------------------------------------------------
// C[M,N] = A[M,K] @ Bt[N,K]^T + bias[N].  A [m][k], Bt [n][k], bf16.
// M,N % 64 == 0, K % 32 == 0.  Block = 128 threads (4 waves), 64x64 tile,
// wave -> 16x64.  LDS double-buffered; staging is contiguous both sides.
template <bool OUT_BF16>
__global__ void gemm_bias_kernel(const unsigned short* __restrict__ A,
                                 const unsigned short* __restrict__ Bt,
                                 const float* __restrict__ bias,
                                 void* __restrict__ Cout,
                                 int M, int N, int K) {
  __shared__ alignas(16) unsigned short ldsA[2][64 * GSTR];   // [m][k]
  __shared__ alignas(16) unsigned short ldsBt[2][64 * GSTR];  // [n][k]

  const int t    = threadIdx.x;
  const int lane = t & 31;
  const int wv   = t >> 5;
  const int l16  = lane & 15;
  const int hh   = lane >> 4;
  const int tn0  = blockIdx.x * 64;
  const int tm0  = blockIdx.y * 64;

  auto stage = [&](int kc, int buf) {
    // A and B tiles: each 64 rows x 32 cols -> 256 x 16B vectors each.
    for (int i = 0; i < 2; ++i) {
      int idx = t + i * 128;
      int r = idx >> 2, seg = (idx & 3) * 8;
      copy16_g2l(&A[(size_t)(tm0 + r) * K + kc + seg],
                 &ldsA[buf][r * GSTR + seg]);
      copy16_g2l(&Bt[(size_t)(tn0 + r) * K + kc + seg],
                 &ldsBt[buf][r * GSTR + seg]);
    }
  };

  v8f acc[4];
  for (int nt = 0; nt < 4; ++nt)
    for (int e = 0; e < 8; ++e) acc[nt][e] = 0.0f;

  const int nk = K >> 5;
  stage(0, 0);
  wait_async_copies();
  for (int kt = 0; kt < nk; ++kt) {
    __syncthreads();
    if (kt + 1 < nk) stage((kt + 1) << 5, (kt + 1) & 1);
    const int buf = kt & 1;

    const FragBF a = load_a_frag(&ldsA[buf][(wv * 16 + l16) * GSTR], hh);
    for (int nt = 0; nt < 4; ++nt) {
      const FragBF b = load_b_frag(&ldsBt[buf][(nt * 16 + l16) * GSTR], hh);
      acc[nt] = __builtin_amdgcn_wmma_f32_16x16x32_bf16(
          false, a.v, false, b.v, (short)0, acc[nt], false, false);
    }
    wait_async_copies();  // our async writes to buf^1 done before next barrier
  }

  // Epilogue: C/D layout -> row = r + 8*half, col = l16 (+tile offsets).
  for (int nt = 0; nt < 4; ++nt) {
    const int col = tn0 + nt * 16 + l16;
    const float bv = bias[col];
    for (int r = 0; r < 8; ++r) {
      const int row = tm0 + wv * 16 + r + hh * 8;
      const float v = acc[nt][r] + bv;
      if constexpr (OUT_BF16)
        ((unsigned short*)Cout)[(size_t)row * N + col] = f2bf(v);
      else
        ((float*)Cout)[(size_t)row * N + col] = v;
    }
  }
}

// ---------------------------------------------------------------------------
// Flash attention: grid = (SEQ/64, BATCH*NHEAD), block = 128 (4 waves).
// Each wave owns 16 query rows; loop over 64-key chunks with online softmax.
__global__ void attn_kernel(const unsigned short* __restrict__ qkv,
                            unsigned short* __restrict__ out) {
  __shared__ alignas(16) unsigned short ldsK[64 * ASTR];      // [key][d]
  __shared__ alignas(16) unsigned short ldsVt[64 * ASTR];     // [d][key]
  __shared__ alignas(16) unsigned short ldsP[4][16 * ASTR];   // per-wave P

  const int t    = threadIdx.x;
  const int lane = t & 31;
  const int wv   = t >> 5;
  const int l16  = lane & 15;
  const int hh   = lane >> 4;
  const int bh   = blockIdx.y;
  const int b    = bh >> 4;
  const int h    = bh & 15;
  const int q0   = blockIdx.x * 64 + wv * 16;     // this wave's first q row
  const size_t rowbase = (size_t)b * SEQ;

  // Q fragments (A operand, contraction over d = 64 -> 2 steps of 32).
  FragBF qf[2];
  {
    const unsigned short* Qp = qkv + (rowbase + q0 + l16) * QKV_N + h * HS;
    qf[0] = load_a_frag(Qp, hh);
    qf[1] = load_a_frag(Qp + 32, hh);
  }

  v8f o[4];
  float rm[8], rl[8];
  for (int nt = 0; nt < 4; ++nt)
    for (int e = 0; e < 8; ++e) o[nt][e] = 0.0f;
  for (int r = 0; r < 8; ++r) { rm[r] = -INFINITY; rl[r] = 0.0f; }

  for (int kv0 = 0; kv0 < SEQ; kv0 += 64) {
    // Stage K [64key][64d] (async b128) and V transposed [64d][64key].
    for (int i = 0; i < 4; ++i) {
      int idx = t + i * 128;                 // 0..511
      int key = idx >> 3, d8 = (idx & 7) * 8;
      size_t g = (rowbase + kv0 + key) * QKV_N + h * HS + d8;
      copy16_g2l(&qkv[g + INNER], &ldsK[key * ASTR + d8]);
      Quad8 vd;
      vd.q = *(const v4u*)&qkv[g + 2 * INNER];
      for (int e = 0; e < 8; ++e) ldsVt[(d8 + e) * ASTR + key] = vd.s[e];
    }
    wait_async_copies();
    __syncthreads();

    // S = (Q K^T) * 1/sqrt(64).  B[kk=d][n=key] = ldsK row `key` (contig in d).
    v8f st[4];
    for (int nt = 0; nt < 4; ++nt) {
      v8f s;
      for (int e = 0; e < 8; ++e) s[e] = 0.0f;
      const unsigned short* kcol = &ldsK[(nt * 16 + l16) * ASTR];
      for (int j = 0; j < 2; ++j) {
        const FragBF kf = load_b_frag(kcol + j * 32, hh);
        s = __builtin_amdgcn_wmma_f32_16x16x32_bf16(
            false, qf[j].v, false, kf.v, (short)0, s, false, false);
      }
      for (int e = 0; e < 8; ++e) s[e] *= 0.125f;
      st[nt] = s;
    }

    // Online softmax per query row r (a row lives in one 16-lane half).
    for (int r = 0; r < 8; ++r) {
      float mx = st[0][r];
      for (int nt = 1; nt < 4; ++nt) mx = fmaxf(mx, st[nt][r]);
      for (int off = 1; off < 16; off <<= 1)
        mx = fmaxf(mx, __shfl_xor(mx, off, 32));
      const float mnew = fmaxf(rm[r], mx);
      const float scale = __expf(rm[r] - mnew);
      rm[r] = mnew;
      float sum = 0.0f;
      for (int nt = 0; nt < 4; ++nt) {
        const float p = __expf(st[nt][r] - mnew);
        sum += p;
        ldsP[wv][(r + hh * 8) * ASTR + nt * 16 + l16] = f2bf(p);
      }
      for (int off = 1; off < 16; off <<= 1)
        sum += __shfl_xor(sum, off, 32);
      rl[r] = rl[r] * scale + sum;
      for (int nt = 0; nt < 4; ++nt) o[nt][r] *= scale;
    }
    // Same-wave LDS ops are in order: no barrier needed before reading ldsP.

    // O += P @ V.  A = P (contraction over keys), B[kk=key][n=d] = ldsVt rows.
    for (int j = 0; j < 2; ++j) {
      const FragBF pf = load_a_frag(&ldsP[wv][l16 * ASTR + j * 32], hh);
      for (int nt = 0; nt < 4; ++nt) {
        const FragBF vf =
            load_b_frag(&ldsVt[(nt * 16 + l16) * ASTR + j * 32], hh);
        o[nt] = __builtin_amdgcn_wmma_f32_16x16x32_bf16(
            false, pf.v, false, vf.v, (short)0, o[nt], false, false);
      }
    }
    __syncthreads();  // before next chunk overwrites ldsK/ldsVt
  }

  // Normalize and emit bf16 attn output (heads re-interleaved to [row][inner]).
  for (int nt = 0; nt < 4; ++nt) {
    for (int r = 0; r < 8; ++r) {
      const float v = o[nt][r] / rl[r];
      const size_t row = rowbase + q0 + r + hh * 8;
      out[row * INNER + h * HS + nt * 16 + l16] = f2bf(v);
    }
  }
}

// ---------------------------------------------------------------------------
extern "C" void kernel_launch(void* const* d_in, const int* in_sizes, int n_in,
                              void* d_out, int out_size, void* d_ws,
                              size_t ws_size, hipStream_t stream) {
  const float* x     = (const float*)d_in[0];
  const float* w_qkv = (const float*)d_in[1];
  const float* b_qkv = (const float*)d_in[2];
  const float* w_out = (const float*)d_in[3];
  const float* b_out = (const float*)d_in[4];

  // Workspace layout (bf16 ushorts): ~88 MB total, all 16B-aligned offsets.
  unsigned short* ws       = (unsigned short*)d_ws;
  unsigned short* x_bf     = ws;                               // 8192*1024
  unsigned short* wqkvT_bf = x_bf + (size_t)ROWS * DIM;        // [3072][1024]
  unsigned short* woutT_bf = wqkvT_bf + (size_t)DIM * QKV_N;   // [1024][1024]
  unsigned short* qkv_bf   = woutT_bf + (size_t)INNER * DIM;   // 8192*3072
  unsigned short* attn_bf  = qkv_bf + (size_t)ROWS * QKV_N;    // 8192*1024

  const int n1 = ROWS * DIM;
  f32_to_bf16_kernel<<<(n1 + 255) / 256, 256, 0, stream>>>(x, x_bf, n1);
  // Weights: convert + transpose once -> [n][k] so GEMM staging is contiguous.
  transpose_to_bf16_kernel<<<dim3(QKV_N / 32, DIM / 32), 256, 0, stream>>>(
      w_qkv, wqkvT_bf, DIM, QKV_N);
  transpose_to_bf16_kernel<<<dim3(DIM / 32, INNER / 32), 256, 0, stream>>>(
      w_out, woutT_bf, INNER, DIM);

  // QKV projection: [8192,1024] @ [1024,3072] + b_qkv -> bf16
  gemm_bias_kernel<true><<<dim3(QKV_N / 64, ROWS / 64), 128, 0, stream>>>(
      x_bf, wqkvT_bf, b_qkv, qkv_bf, ROWS, QKV_N, DIM);

  // Flash attention per (b,h), 64 query rows per block.
  attn_kernel<<<dim3(SEQ / 64, BATCH * NHEAD), 128, 0, stream>>>(qkv_bf,
                                                                 attn_bf);

  // Output projection: [8192,1024] @ [1024,1024] + b_out -> f32 d_out
  gemm_bias_kernel<false><<<dim3(DIM / 64, ROWS / 64), 128, 0, stream>>>(
      attn_bf, woutT_bf, b_out, d_out, ROWS, DIM, INNER);
}
